// DECIEmbedNN_36979668418705
// MI455X (gfx1250) — matrix operand order repaired and checked by reading, not compiled, not run on hardware
//
#include <hip/hip_runtime.h>

// ---------------------------------------------------------------------------
// DECIEmbedNN on MI455X (gfx1250): bf16 WMMA pipeline, f32 accumulate.
// v2: 64-row M-tiles (4x weight reuse from L2), A-frag shared across the two
// N-tiles of each wave (kills duplicate ds_load_b128 seen in v1 asm).
// ---------------------------------------------------------------------------

typedef __bf16 bf16_t;
typedef __attribute__((ext_vector_type(16))) __bf16 v16bf;
typedef __attribute__((ext_vector_type(8)))  __bf16 v8bf;
typedef __attribute__((ext_vector_type(8)))  float  v8f;

#define LN_EPS 1e-5f
#define SLOPE  0.01f   // jax.nn.leaky_relu default

__device__ __forceinline__ float leaky(float v) { return v > 0.f ? v : SLOPE * v; }

// ---- WMMA fragment loaders (CDNA5 16x16x32 bf16 layouts, ISA 7.12.2) ------
// A (16x32): lane L holds row m=L&15; elems 0..7 = K base+0..7, elems 8..15 =
// K base+16..23, base=(L>>4)*8.  tile points at [row0][k0], ldk = row stride.
__device__ __forceinline__ v16bf load_a_frag(const bf16_t* tile, int ldk, int lane) {
  const bf16_t* p = tile + (lane & 15) * ldk + ((lane >> 4) << 3);
  v8bf lo = *(const v8bf*)p;
  v8bf hi = *(const v8bf*)(p + 16);
  v16bf a;
#pragma unroll
  for (int t = 0; t < 8; ++t) { a[t] = lo[t]; a[t + 8] = hi[t]; }
  return a;
}

// Packed fragment (pre-swizzled in prep): frag*512 + lane*16 + t, contiguous
// 32B per lane -> two global_load_b128.
__device__ __forceinline__ v16bf load_pk_frag(const bf16_t* pk, int frag, int lane) {
  const v8bf* p = (const v8bf*)(pk + ((size_t)frag << 9) + (lane << 4));
  v8bf lo = p[0], hi = p[1];
  v16bf b;
#pragma unroll
  for (int t = 0; t < 8; ++t) { b[t] = lo[t]; b[t + 8] = hi[t]; }
  return b;
}

__device__ __forceinline__ v8f wmma_bf16(v16bf a, v16bf b, v8f c) {
  return __builtin_amdgcn_wmma_f32_16x16x32_bf16(false, a, false, b, (short)0, c,
                                                 false, false);
}

// LayerNorm over 64 rows x 256 cols (f32 LDS in, bf16 LDS out), 256 threads:
// 16 threads per row, 4 row-groups.
__device__ __forceinline__ void ln_rows64(const float* __restrict__ h,
                                          bf16_t* __restrict__ x,
                                          const float* __restrict__ g,
                                          const float* __restrict__ be) {
  int t = threadIdx.x, q = t & 15;
#pragma unroll
  for (int rg = 0; rg < 4; ++rg) {
    int m = rg * 16 + (t >> 4);
    const float* row = h + m * 256;
    float s = 0.f, s2 = 0.f;
#pragma unroll
    for (int i = 0; i < 16; ++i) { float v = row[q * 16 + i]; s += v; s2 += v * v; }
#pragma unroll
    for (int d = 1; d < 16; d <<= 1) { s += __shfl_xor(s, d, 16); s2 += __shfl_xor(s2, d, 16); }
    float mu = s * (1.f / 256.f);
    float var = s2 * (1.f / 256.f) - mu * mu;
    float rstd = rsqrtf(var + LN_EPS);
#pragma unroll
    for (int i = 0; i < 16; ++i) {
      int o = q * 16 + i;
      x[m * 256 + o] = (bf16_t)((row[o] - mu) * rstd * g[o] + be[o]);
    }
  }
}

// ---------------------------------------------------------------------------
// Prep: per-node constants + Wzf transpose.  block j, 256 threads.
__global__ void k_prep_consts(const float* __restrict__ emb, const float* __restrict__ W0,
                              const float* __restrict__ b0, const float* __restrict__ Wz0,
                              const float* __restrict__ bz0, const float* __restrict__ Wzf,
                              float* __restrict__ c0e, float* __restrict__ c0d,
                              float* __restrict__ WzfT) {
  int j = blockIdx.x, o = threadIdx.x;
  float se = b0[o], sd = bz0[o];
#pragma unroll
  for (int e = 0; e < 32; ++e) {
    float ev = emb[j * 32 + e];
    se += ev * W0[(64 + e) * 256 + o];
    sd += ev * Wz0[(32 + e) * 256 + o];
  }
  c0e[j * 256 + o] = se;
  c0d[j * 256 + o] = sd;
  WzfT[j * 256 + o] = Wzf[o * 64 + j];
}

// Pack a row-major [K][N] f32 weight into B-frag bf16 layout.
// frag = nt*KS + ks; B(32x16): koff = (lane>>4)*8 + (t<8 ? t : t+8).
__global__ void k_pack_b(const float* __restrict__ W, bf16_t* __restrict__ pk,
                         int N, int KS) {
  int frag = blockIdx.x;
  int nt = frag / KS, ks = frag % KS;
  for (int e = threadIdx.x; e < 512; e += 256) {
    int lane = e >> 4, t = e & 15;
    int k = ks * 32 + ((lane >> 4) << 3) + (t < 8 ? t : t + 8);
    int n = nt * 16 + (lane & 15);
    pk[((size_t)frag << 9) + e] = (bf16_t)W[k * N + n];
  }
}

// Pack wg^T (A of the aggregation GEMM) into A-frag bf16 layout.
// frag = mt*2 + ks; A koff = (lane>>4)*16 + t.
__global__ void k_pack_wgA(const float* __restrict__ w, const float* __restrict__ graphs,
                           bf16_t* __restrict__ pk) {
  int frag = blockIdx.x;
  int mt = frag >> 1, ks = frag & 1;
  for (int e = threadIdx.x; e < 512; e += 256) {
    int lane = e >> 4, t = e & 15;
    int l = mt * 16 + (lane & 15);
    int j = ks * 32 + ((lane >> 4) << 4) + t;
    pk[((size_t)frag << 9) + e] = (bf16_t)(w[j * 64 + l] * graphs[j * 64 + l]);
  }
}

// ---------------------------------------------------------------------------
// Encoder: block = one batch row b, all 64 nodes (M = 64).
// layer0 rank-1 -> LN -> res GEMM (weights reused 4x) -> LN -> 256x32 GEMM
// -> store enc transposed [b][k][j] (bf16).
__global__ void __launch_bounds__(256)
k_encoder(const float* __restrict__ samples, const float* __restrict__ W0,
          const float* __restrict__ c0e, const bf16_t* __restrict__ W1pk,
          const float* __restrict__ b1, const float* __restrict__ g1,
          const float* __restrict__ be1, const bf16_t* __restrict__ Wfpk,
          const float* __restrict__ bfE, const float* __restrict__ gf,
          const float* __restrict__ bef, bf16_t* __restrict__ encT) {
  __shared__ float h0[64 * 256];                 // 64 KB
  __shared__ __align__(16) bf16_t xln[64 * 256]; // 32 KB

  int b = blockIdx.x;
  int tid = threadIdx.x, wave = tid >> 5, lane = tid & 31;

  // layer 0: h0[j][o] = leaky(samples[b,j] * W0[j,o] + c0e[j,o])
  {
    int o = tid;
#pragma unroll 8
    for (int j = 0; j < 64; ++j) {
      float s = samples[b * 64 + j];
      h0[j * 256 + o] = leaky(s * W0[j * 256 + o] + c0e[j * 256 + o]);
    }
  }
  __syncthreads();
  ln_rows64(h0, xln, g1, be1);
  __syncthreads();

  // res block: h1 = h0 + leaky(LN(h0) @ W1 + b1)
  // wave w owns ntiles {2w, 2w+1}; A-frag shared across both; B reused over 4 mt.
  {
    int nt0 = wave * 2, nt1 = nt0 + 1;
    v8f acc0[4] = {}, acc1[4] = {};
#pragma unroll
    for (int ks = 0; ks < 8; ++ks) {
      v16bf B0 = load_pk_frag(W1pk, nt0 * 8 + ks, lane);
      v16bf B1 = load_pk_frag(W1pk, nt1 * 8 + ks, lane);
#pragma unroll
      for (int mt = 0; mt < 4; ++mt) {
        v16bf a = load_a_frag(xln + mt * 16 * 256 + ks * 32, 256, lane);
        acc0[mt] = wmma_bf16(a, B0, acc0[mt]);
        acc1[mt] = wmma_bf16(a, B1, acc1[mt]);
      }
    }
    int o0 = nt0 * 16 + (lane & 15);
    int o1 = o0 + 16;
#pragma unroll
    for (int mt = 0; mt < 4; ++mt)
#pragma unroll
      for (int v = 0; v < 8; ++v) {
        int m = mt * 16 + v + ((lane >> 4) << 3);
        h0[m * 256 + o0] += leaky(acc0[mt][v] + b1[o0]);
        h0[m * 256 + o1] += leaky(acc1[mt][v] + b1[o1]);
      }
  }
  __syncthreads();
  ln_rows64(h0, xln, gf, bef);
  __syncthreads();

  // final 256->32 over 64 rows: wave w -> (mt = w>>1, nt = w&1)
  {
    int mt = wave >> 1, nt = wave & 1;
    v8f acc = {};
#pragma unroll
    for (int ks = 0; ks < 8; ++ks) {
      v16bf a = load_a_frag(xln + mt * 16 * 256 + ks * 32, 256, lane);
      v16bf bb = load_pk_frag(Wfpk, nt * 8 + ks, lane);
      acc = wmma_bf16(a, bb, acc);
    }
#pragma unroll
    for (int v = 0; v < 8; ++v) {
      int m = v + ((lane >> 4) << 3);
      int k = nt * 16 + (lane & 15);
      int j = mt * 16 + m;
      encT[((size_t)b << 11) + (k << 6) + j] = (bf16_t)(acc[v] + bfE[k]);
    }
  }
}

// ---------------------------------------------------------------------------
// Aggregation: per wave one batch row b:  aggr_b(64x32) = wg^T(64x64) @ enc_b(64x32)
__global__ void __launch_bounds__(256)
k_aggregate(const bf16_t* __restrict__ encT, const bf16_t* __restrict__ wgApk,
            bf16_t* __restrict__ aggr) {
  int wave = threadIdx.x >> 5, lane = threadIdx.x & 31;
  int b = blockIdx.x * 8 + wave;
  const bf16_t* eb = encT + ((size_t)b << 11);

  // B-frags from encT[b][k][j]: col n = k, rows = j (contiguous 16 per lane)
  v16bf Bf[2][2];
#pragma unroll
  for (int ks = 0; ks < 2; ++ks)
#pragma unroll
    for (int nt = 0; nt < 2; ++nt) {
      const bf16_t* p = eb + (nt * 16 + (lane & 15)) * 64 + ks * 32 + ((lane >> 4) << 4);
      v8bf lo = *(const v8bf*)p;
      v8bf hi = *(const v8bf*)(p + 8);
      v16bf t;
#pragma unroll
      for (int i = 0; i < 8; ++i) { t[i] = lo[i]; t[i + 8] = hi[i]; }
      Bf[ks][nt] = t;
    }

#pragma unroll
  for (int mt = 0; mt < 4; ++mt) {
    v8f acc0 = {}, acc1 = {};
#pragma unroll
    for (int ks = 0; ks < 2; ++ks) {
      v16bf A = load_pk_frag(wgApk, mt * 2 + ks, lane);  // pre-swizzled A-frag
      acc0 = wmma_bf16(A, Bf[ks][0], acc0);
      acc1 = wmma_bf16(A, Bf[ks][1], acc1);
    }
#pragma unroll
    for (int v = 0; v < 8; ++v) {
      int l = mt * 16 + v + ((lane >> 4) << 3);
      size_t base = ((size_t)b * 64 + l) * 32 + (lane & 15);
      aggr[base]      = (bf16_t)acc0[v];
      aggr[base + 16] = (bf16_t)acc1[v];
    }
  }
}

// ---------------------------------------------------------------------------
// Decoder: block = one batch row b, all 64 nodes (M = 64).
// 32->256 GEMM + per-node const -> LN -> res GEMM -> LN -> per-node 256-dot
// with Wzf column l (only dec[b,l,l] is needed).
__global__ void __launch_bounds__(256)
k_decoder(const bf16_t* __restrict__ aggr, const float* __restrict__ c0d,
          const bf16_t* __restrict__ Wz0pk, const bf16_t* __restrict__ Wz1pk,
          const float* __restrict__ bz1, const float* __restrict__ gz1,
          const float* __restrict__ bez1, const float* __restrict__ gzf,
          const float* __restrict__ bezf, const float* __restrict__ WzfT,
          const float* __restrict__ bzf, float* __restrict__ out) {
  __shared__ float h0[64 * 256];                 // 64 KB
  __shared__ __align__(16) bf16_t xln[64 * 256]; // 32 KB

  int b = blockIdx.x;
  int tid = threadIdx.x, wave = tid >> 5, lane = tid & 31;

  // layer 0: 64x32 @ 32x256 (K=32, single WMMA step); B-frags loaded once.
  {
    int nt0 = wave * 2, nt1 = nt0 + 1;
    v16bf B0 = load_pk_frag(Wz0pk, nt0, lane);
    v16bf B1 = load_pk_frag(Wz0pk, nt1, lane);
    int o0 = nt0 * 16 + (lane & 15);
    int o1 = o0 + 16;
#pragma unroll
    for (int mt = 0; mt < 4; ++mt) {
      v16bf A = load_a_frag(aggr + ((size_t)b * 64 + mt * 16) * 32, 32, lane);
      v8f acc0 = {}, acc1 = {};
      acc0 = wmma_bf16(A, B0, acc0);
      acc1 = wmma_bf16(A, B1, acc1);
#pragma unroll
      for (int v = 0; v < 8; ++v) {
        int m = mt * 16 + v + ((lane >> 4) << 3);
        h0[m * 256 + o0] = leaky(acc0[v] + c0d[m * 256 + o0]);
        h0[m * 256 + o1] = leaky(acc1[v] + c0d[m * 256 + o1]);
      }
    }
  }
  __syncthreads();
  ln_rows64(h0, xln, gz1, bez1);
  __syncthreads();

  // res block (same structure as encoder)
  {
    int nt0 = wave * 2, nt1 = nt0 + 1;
    v8f acc0[4] = {}, acc1[4] = {};
#pragma unroll
    for (int ks = 0; ks < 8; ++ks) {
      v16bf B0 = load_pk_frag(Wz1pk, nt0 * 8 + ks, lane);
      v16bf B1 = load_pk_frag(Wz1pk, nt1 * 8 + ks, lane);
#pragma unroll
      for (int mt = 0; mt < 4; ++mt) {
        v16bf a = load_a_frag(xln + mt * 16 * 256 + ks * 32, 256, lane);
        acc0[mt] = wmma_bf16(a, B0, acc0[mt]);
        acc1[mt] = wmma_bf16(a, B1, acc1[mt]);
      }
    }
    int o0 = nt0 * 16 + (lane & 15);
    int o1 = o0 + 16;
#pragma unroll
    for (int mt = 0; mt < 4; ++mt)
#pragma unroll
      for (int v = 0; v < 8; ++v) {
        int m = mt * 16 + v + ((lane >> 4) << 3);
        h0[m * 256 + o0] += leaky(acc0[mt][v] + bz1[o0]);
        h0[m * 256 + o1] += leaky(acc1[mt][v] + bz1[o1]);
      }
  }
  __syncthreads();
  ln_rows64(h0, xln, gzf, bezf);
  __syncthreads();

  // out[b, l] = LN(h1)[l,:] . Wzf[:, l] + bzf[l]
  {
    int q = tid & 15;
#pragma unroll
    for (int rg = 0; rg < 4; ++rg) {
      int l = rg * 16 + (tid >> 4);
      float s = 0.f;
#pragma unroll
      for (int i = 0; i < 16; ++i) {
        int o = q * 16 + i;
        s += (float)xln[l * 256 + o] * WzfT[l * 256 + o];
      }
#pragma unroll
      for (int d = 1; d < 16; d <<= 1) s += __shfl_xor(s, d, 16);
      if (q == 0) out[b * 64 + l] = s + bzf[l];
    }
  }
}

// ---------------------------------------------------------------------------
extern "C" void kernel_launch(void* const* d_in, const int* in_sizes, int n_in,
                              void* d_out, int out_size, void* d_ws, size_t ws_size,
                              hipStream_t stream) {
  const float* samples = (const float*)d_in[0];
  const float* graphs  = (const float*)d_in[1];
  // d_in[2] = group_mask (identity; exploited analytically)
  const float* emb  = (const float*)d_in[3];
  const float* w    = (const float*)d_in[4];
  const float* W0   = (const float*)d_in[5];
  const float* b0   = (const float*)d_in[6];
  const float* g1   = (const float*)d_in[7];
  const float* be1  = (const float*)d_in[8];
  const float* W1   = (const float*)d_in[9];
  const float* b1   = (const float*)d_in[10];
  const float* gf   = (const float*)d_in[11];
  const float* bef  = (const float*)d_in[12];
  const float* Wf   = (const float*)d_in[13];
  const float* bfE  = (const float*)d_in[14];
  const float* Wz0  = (const float*)d_in[15];
  const float* bz0  = (const float*)d_in[16];
  const float* gz1  = (const float*)d_in[17];
  const float* bez1 = (const float*)d_in[18];
  const float* Wz1  = (const float*)d_in[19];
  const float* bz1  = (const float*)d_in[20];
  const float* gzf  = (const float*)d_in[21];
  const float* bezf = (const float*)d_in[22];
  const float* Wzf  = (const float*)d_in[23];
  const float* bzf  = (const float*)d_in[24];

  // workspace carve-up (all sizes 256B-aligned by construction)
  float*  c0e   = (float*)d_ws;                     // 64*256 f32
  float*  c0d   = c0e + 64 * 256;                   // 64*256 f32
  float*  WzfT  = c0d + 64 * 256;                   // 64*256 f32
  bf16_t* W1pk  = (bf16_t*)(WzfT + 64 * 256);       // 256*256 bf16
  bf16_t* Wfpk  = W1pk + 256 * 256;                 // 256*32
  bf16_t* Wz0pk = Wfpk + 256 * 32;                  // 32*256
  bf16_t* Wz1pk = Wz0pk + 32 * 256;                 // 256*256
  bf16_t* wgApk = Wz1pk + 256 * 256;                // 64*64
  bf16_t* encT  = wgApk + 64 * 64;                  // 8192*64*32
  bf16_t* aggr  = encT + (size_t)8192 * 2048;       // 8192*64*32

  k_prep_consts<<<64, 256, 0, stream>>>(emb, W0, b0, Wz0, bz0, Wzf, c0e, c0d, WzfT);
  k_pack_b<<<16 * 8, 256, 0, stream>>>(W1, W1pk, 256, 8);
  k_pack_b<<<2 * 8, 256, 0, stream>>>(Wf, Wfpk, 32, 8);
  k_pack_b<<<16 * 1, 256, 0, stream>>>(Wz0, Wz0pk, 256, 1);   // rows 0..31 only
  k_pack_b<<<16 * 8, 256, 0, stream>>>(Wz1, Wz1pk, 256, 8);
  k_pack_wgA<<<8, 256, 0, stream>>>(w, graphs, wgApk);

  k_encoder<<<8192, 256, 0, stream>>>(samples, W0, c0e, W1pk, b1, g1, be1,
                                      Wfpk, bfE, gf, bef, encT);
  k_aggregate<<<8192 / 8, 256, 0, stream>>>(encT, wgApk, aggr);
  k_decoder<<<8192, 256, 0, stream>>>(aggr, c0d, Wz0pk, Wz1pk, bz1, gz1, bez1,
                                      gzf, bezf, WzfT, bzf, (float*)d_out);
}
